// TopologicalAttention_29111288333012
// MI455X (gfx1250) — matrix operand order repaired and checked by reading, not compiled, not run on hardware
//
#include <hip/hip_runtime.h>

typedef __bf16 bf16_t;
typedef __attribute__((ext_vector_type(16))) __bf16 v16bf;
typedef __attribute__((ext_vector_type(8)))  float  v8f;

union Frag16 { v16bf v; uint4 q[2]; };

static __device__ __forceinline__ v8f wmma_bf16(const Frag16& a, const Frag16& b, v8f c) {
    return __builtin_amdgcn_wmma_f32_16x16x32_bf16(false, a.v, false, b.v, (short)0, c, false, false);
}

#define BB      2
#define SS      2048
#define DD      768
#define HH      12
#define HDIM    64
#define TOPO_   128
#define NHASH   8
#define HBITS   64
#define DAUG    896      // D + TOPO
#define MTOT    4096     // B*S
#define NQKV    2304     // 3*D
#define NHB     512      // NHASH*HBITS
#define KWORDS  64       // S/32
#define CWORDS  16       // 512 bits / 32
#define NEGV    (-10000.0f)

// ---------------------------------------------------------------------------
// Elementwise prep kernels
// ---------------------------------------------------------------------------
__global__ void prep_x_kernel(const float* __restrict__ hs, const float* __restrict__ tf,
                              bf16_t* __restrict__ Xa, bf16_t* __restrict__ Ca) {
    int i = blockIdx.x * blockDim.x + threadIdx.x;
    if (i >= MTOT * DAUG) return;
    int c = i % DAUG, row = i / DAUG;
    if (c < DD) {
        Xa[i] = (bf16_t)hs[(size_t)row * DD + c];
    } else {
        bf16_t v = (bf16_t)tf[(size_t)row * TOPO_ + (c - DD)];
        Xa[i] = v;
        Ca[i] = v;           // context cols [0,768) written later by attention
    }
}

__global__ void pack_wqkv_kernel(const float* __restrict__ Wq, const float* __restrict__ Wk,
                                 const float* __restrict__ Wv, bf16_t* __restrict__ WT) {
    int i = blockIdx.x * blockDim.x + threadIdx.x;
    if (i >= NQKV * DD) return;
    int k = i % DD, n = i / DD;
    const float* W; int nn;
    if (n < DD)            { W = Wq; nn = n; }
    else if (n < 2 * DD)   { W = Wk; nn = n - DD; }
    else                   { W = Wv; nn = n - 2 * DD; }
    WT[i] = (bf16_t)W[(size_t)k * DD + nn];
}

__global__ void pack_whash_kernel(const float* __restrict__ hp, bf16_t* __restrict__ WT) {
    // WT[n][k], n = h*64 + c over 512, k over 896.  TEMP=0.1 folded into topo rows.
    int i = blockIdx.x * blockDim.x + threadIdx.x;
    if (i >= NHB * DAUG) return;
    int k = i % DAUG, n = i / DAUG;
    int h = n >> 6, c = n & 63;
    float v = (k < DD) ? hp[((size_t)h * DD + k) * HBITS + c]
                       : 0.1f * hp[((size_t)h * DD + (k - DD)) * HBITS + c];
    WT[i] = (bf16_t)v;
}

__global__ void pack_wt_kernel(const float* __restrict__ W, bf16_t* __restrict__ WT,
                               int N, int Kd) {
    // W is [Kd][N] row-major  ->  WT is [N][Kd]
    int i = blockIdx.x * blockDim.x + threadIdx.x;
    if (i >= N * Kd) return;
    int k = i % Kd, n = i / Kd;
    WT[i] = (bf16_t)W[(size_t)k * N + n];
}

__global__ void pack_codes_kernel(const float* __restrict__ proj, unsigned* __restrict__ codes) {
    int i = blockIdx.x * blockDim.x + threadIdx.x;
    if (i >= MTOT * CWORDS) return;
    int w = i % CWORDS, t = i / CWORDS;
    const float* p = proj + (size_t)t * NHB + w * 32;
    unsigned o = 0u;
#pragma unroll
    for (int j = 0; j < 32; ++j) o |= (p[j] >= 0.0f ? 1u : 0u) << j;
    codes[i] = o;
}

__global__ void build_mask_kernel(const unsigned* __restrict__ codes,
                                  const int* __restrict__ amask,
                                  unsigned* __restrict__ keepw) {
    // keep iff  sim = (512 - 2*popcnt(xor)) / 512 > 0.3  <=>  popcnt <= 179,  AND pad mask
    int i = blockIdx.x * blockDim.x + threadIdx.x;
    if (i >= BB * SS * KWORDS) return;
    int w = i % KWORDS;
    int q = (i / KWORDS) % SS;
    int b = i / (KWORDS * SS);
    unsigned cq[CWORDS];
    const unsigned* cqp = codes + ((size_t)(b * SS + q)) * CWORDS;
#pragma unroll
    for (int t = 0; t < CWORDS; ++t) cq[t] = cqp[t];
    unsigned out = 0u;
    for (int j = 0; j < 32; ++j) {
        int k = w * 32 + j;
        const unsigned* ck = codes + ((size_t)(b * SS + k)) * CWORDS;
        int p = 0;
#pragma unroll
        for (int t = 0; t < CWORDS; ++t) p += __popc(cq[t] ^ ck[t]);
        bool keep = (p <= 179) && (amask[b * SS + k] != 0);
        out |= (keep ? 1u : 0u) << j;
    }
    keepw[i] = out;
}

// ---------------------------------------------------------------------------
// Generic bf16 WMMA GEMM:  C[M x N] = A[M x K] * BT[N x K]^T,  tile 128x64,
// 256 threads = 8 wave32, each wave owns a 16x64 strip (4 WMMA n-tiles).
// LDS double-buffered; globals for tile i+1 staged in registers while tile i
// runs on the matrix pipe.  Main loop is peeled so the steady state has an
// unconditional prefetch and a single barrier per k-step.
// mode 0: store bf16 (QKV)        mode 1: store f32 + bias (hash proj)
// mode 2: gate epilogue -> bf16   mode 3: store f32 + bias (final output)
// ---------------------------------------------------------------------------
__global__ __launch_bounds__(256)
void gemm_bf16_wmma(const bf16_t* __restrict__ A, int lda,
                    const bf16_t* __restrict__ BT, int ldb,
                    int K, int mode,
                    const float* __restrict__ bias,
                    const bf16_t* __restrict__ gc,   // mode2: context (stride DAUG)
                    const float* __restrict__ gh,    // mode2: hidden_states (stride DD)
                    bf16_t* __restrict__ outb, int ldob,
                    float* __restrict__ outf, int ldof) {
    __shared__ bf16_t As[2][128 * 40];
    __shared__ bf16_t Bs[2][64 * 40];

    const int tid = threadIdx.x;
    const int wv = tid >> 5, ln = tid & 31, r = ln & 15, hf = ln >> 4;
    const int tileM = blockIdx.x * 128, tileN = blockIdx.y * 64;

    v8f acc[4] = {};

    const int arow = tid >> 1, akoff = (tid & 1) * 16;   // 128 rows x 32 k, 16 elems/thread
    const int brow = tid >> 2, bkoff = (tid & 3) * 8;    // 64 rows x 32 k,   8 elems/thread
    const bf16_t* aptr = A  + (size_t)(tileM + arow) * lda + akoff;
    const bf16_t* bptr = BT + (size_t)(tileN + brow) * ldb + bkoff;
    const int awoff = arow * 40 + akoff;                 // LDS write offsets
    const int bwoff = brow * 40 + bkoff;
    const int aroff = (wv * 16 + r) * 40 + hf * 8;       // LDS read offsets
    const int broff = r * 40 + hf * 8;

    // one k-step of WMMA work against buffer `bi`
    auto compute = [&](int bi) {
        Frag16 af;   // A: lane r holds row r; K split {hf*8..+7, 16+hf*8..+7}
        const bf16_t* ab = As[bi] + aroff;
        af.q[0] = *(const uint4*)(ab);
        af.q[1] = *(const uint4*)(ab + 16);
        Frag16 bfr[4];
#pragma unroll
        for (int nt = 0; nt < 4; ++nt) {
            const bf16_t* bb = Bs[bi] + nt * 16 * 40 + broff;
            bfr[nt].q[0] = *(const uint4*)(bb);
            bfr[nt].q[1] = *(const uint4*)(bb + 16);
        }
#pragma unroll
        for (int nt = 0; nt < 4; ++nt)
            acc[nt] = wmma_bf16(af, bfr[nt], acc[nt]);
    };

    // stage tile 0 and park it in buffer 0
    {
        uint4 a0 = *(const uint4*)(aptr);
        uint4 a1 = *(const uint4*)(aptr + 8);
        uint4 b0 = *(const uint4*)(bptr);
        uint4* d0 = (uint4*)(As[0] + awoff);
        d0[0] = a0; d0[1] = a1;
        *(uint4*)(Bs[0] + bwoff) = b0;
    }
    __syncthreads();

    const int nk = K >> 5;
    int buf = 0;
    for (int it = 0; it < nk - 1; ++it) {     // steady state: unconditional prefetch
        const int kb = (it + 1) << 5;
        uint4 a0 = *(const uint4*)(aptr + kb);
        uint4 a1 = *(const uint4*)(aptr + kb + 8);
        uint4 b0 = *(const uint4*)(bptr + kb);

        compute(buf);

        const int nb = buf ^ 1;
        uint4* d0 = (uint4*)(As[nb] + awoff);
        d0[0] = a0; d0[1] = a1;
        *(uint4*)(Bs[nb] + bwoff) = b0;
        buf = nb;
        __syncthreads();
    }
    compute(buf);                              // peeled last k-step

#pragma unroll
    for (int nt = 0; nt < 4; ++nt) {
#pragma unroll
        for (int j = 0; j < 8; ++j) {
            const int row = tileM + wv * 16 + hf * 8 + j;   // D layout: m = j + 8*half
            const int n = tileN + nt * 16 + r;              //           n = lane&15
            float v = acc[nt][j];
            if (mode == 0) {
                outb[(size_t)row * ldob + n] = (bf16_t)v;
            } else if (mode == 1) {
                outf[(size_t)row * ldof + n] = v + bias[n];
            } else if (mode == 2) {
                float g = 1.0f / (1.0f + __expf(-(v + bias[n])));
                float c = (float)gc[(size_t)row * DAUG + n];
                float h = gh[(size_t)row * DD + n];
                outb[(size_t)row * ldob + n] = (bf16_t)(g * c + (1.0f - g) * h);
            } else {
                outf[(size_t)row * ldof + n] = v + bias[n];
            }
        }
    }
}

// ---------------------------------------------------------------------------
// Flash attention with LSH bitmask.  One block = (b, h, 64-query tile),
// 128 threads = 4 wave32, each wave owns 16 query rows.  QK^T and PV on WMMA,
// online softmax, mask from precomputed keep-bit words.  K/V tiles are staged
// in registers: next tile's global loads issue right after the post-store
// barrier and retire under QK^T + softmax.
// ---------------------------------------------------------------------------
__global__ __launch_bounds__(128)
void attn_kernel(const bf16_t* __restrict__ QKV,
                 const unsigned* __restrict__ keepw,
                 bf16_t* __restrict__ Ca) {
    __shared__ bf16_t Qs [64 * 72];
    __shared__ bf16_t Ks [64 * 72];
    __shared__ bf16_t Vts[64 * 72];       // V transposed: [hd][key]
    __shared__ bf16_t Ps [4 * 16 * 72];   // per-wave P tile [16 x 64]

    const int tid = threadIdx.x, wv = tid >> 5, ln = tid & 31, r = ln & 15, hf = ln >> 4;
    const int qt = blockIdx.x, h = blockIdx.y, b = blockIdx.z;
    const int qbase = qt * 64;
    const int lrow = tid >> 1, coff = (tid & 1) * 32;

    { // load Q tile (64 x 64)
        const uint4* src = (const uint4*)(QKV + ((size_t)(b * SS + qbase + lrow)) * NQKV + h * HDIM + coff);
        uint4* dst = (uint4*)(Qs + lrow * 72 + coff);
        dst[0] = src[0]; dst[1] = src[1]; dst[2] = src[2]; dst[3] = src[3];
    }
    __syncthreads();
    Frag16 qf[2];
#pragma unroll
    for (int kk = 0; kk < 2; ++kk) {
        const bf16_t* base = Qs + (wv * 16 + r) * 72 + kk * 32 + hf * 8;
        qf[kk].q[0] = *(const uint4*)base;
        qf[kk].q[1] = *(const uint4*)(base + 16);
    }

    v8f O[4] = {};
    float mrun[8], lrun[8];
#pragma unroll
    for (int j = 0; j < 8; ++j) { mrun[j] = -1e30f; lrun[j] = 0.0f; }
    const float invscale = 0.125f;       // 1/sqrt(64)
    const unsigned* kwb = keepw + (size_t)b * SS * KWORDS;

    // register-staged K/V tiles
    uint4 krg[4], vrg[4];
    const size_t kvrow = (size_t)(b * SS + lrow) * NQKV + h * HDIM + coff;
    const bf16_t* kbase = QKV + kvrow + DD;
    const bf16_t* vbase = QKV + kvrow + 2 * DD;
    {
        const uint4* ks = (const uint4*)(kbase);
        krg[0] = ks[0]; krg[1] = ks[1]; krg[2] = ks[2]; krg[3] = ks[3];
        const uint4* vs = (const uint4*)(vbase);
        vrg[0] = vs[0]; vrg[1] = vs[1]; vrg[2] = vs[2]; vrg[3] = vs[3];
    }

    for (int kt = 0; kt < SS; kt += 64) {
        { // park staged K/V into LDS (V transposed)
            uint4* kd = (uint4*)(Ks + lrow * 72 + coff);
            kd[0] = krg[0]; kd[1] = krg[1]; kd[2] = krg[2]; kd[3] = krg[3];
            const bf16_t* ve = (const bf16_t*)vrg;
#pragma unroll
            for (int c = 0; c < 32; ++c) Vts[(coff + c) * 72 + lrow] = ve[c];
        }
        __syncthreads();

        // stage next K/V tile: loads retire under QK^T + softmax
        if (kt + 64 < SS) {
            const size_t adv = (size_t)(kt + 64) * NQKV;
            const uint4* ks = (const uint4*)(kbase + adv);
            krg[0] = ks[0]; krg[1] = ks[1]; krg[2] = ks[2]; krg[3] = ks[3];
            const uint4* vs = (const uint4*)(vbase + adv);
            vrg[0] = vs[0]; vrg[1] = vs[1]; vrg[2] = vs[2]; vrg[3] = vs[3];
        }

        // S = Q * K^T  (B-matrix rows = key rows of K, contiguous);
        // load all 4 K frags, then issue the 4 WMMAs back-to-back
        v8f s[4] = {};
#pragma unroll
        for (int kk = 0; kk < 2; ++kk) {
            Frag16 kf[4];
#pragma unroll
            for (int nt = 0; nt < 4; ++nt) {
                const bf16_t* kb2 = Ks + (nt * 16 + r) * 72 + kk * 32 + hf * 8;
                kf[nt].q[0] = *(const uint4*)kb2;
                kf[nt].q[1] = *(const uint4*)(kb2 + 16);
            }
#pragma unroll
            for (int nt = 0; nt < 4; ++nt)
                s[nt] = wmma_bf16(qf[kk], kf[nt], s[nt]);
        }

        // mask + online softmax; rows j live in lane-halves, reduce across 16 lanes
        const int w0 = kt >> 5;
#pragma unroll
        for (int j = 0; j < 8; ++j) {
            const int q = qbase + wv * 16 + hf * 8 + j;
            const unsigned* kw = kwb + (size_t)q * KWORDS + w0;
            const unsigned kw0 = kw[0], kw1 = kw[1];
            float sv[4]; float mloc = -1e30f;
#pragma unroll
            for (int nt = 0; nt < 4; ++nt) {
                int kl = nt * 16 + r;
                unsigned word = (kl < 32) ? kw0 : kw1;
                float x = s[nt][j] * invscale;
                if (!((word >> (kl & 31)) & 1u)) x = NEGV;
                sv[nt] = x; mloc = fmaxf(mloc, x);
            }
#pragma unroll
            for (int d = 1; d < 16; d <<= 1) mloc = fmaxf(mloc, __shfl_xor(mloc, d, 32));
            float mnew = fmaxf(mrun[j], mloc);
            float corr = __expf(mrun[j] - mnew);
            float lloc = 0.0f;
#pragma unroll
            for (int nt = 0; nt < 4; ++nt) {
                float p = __expf(sv[nt] - mnew);
                lloc += p;
                Ps[wv * 16 * 72 + (hf * 8 + j) * 72 + nt * 16 + r] = (bf16_t)p;
            }
#pragma unroll
            for (int d = 1; d < 16; d <<= 1) lloc += __shfl_xor(lloc, d, 32);
            lrun[j] = lrun[j] * corr + lloc;
            mrun[j] = mnew;
#pragma unroll
            for (int nt = 0; nt < 4; ++nt) O[nt][j] *= corr;
        }
        __syncthreads();

        // O += P * V   (A = per-wave P tile, B = V^T rows = hd rows of Vts);
        // grouped frag loads, then 4 WMMAs back-to-back per k-step
#pragma unroll
        for (int kk = 0; kk < 2; ++kk) {
            Frag16 pf;
            const bf16_t* pb = Ps + wv * 16 * 72 + r * 72 + kk * 32 + hf * 8;
            pf.q[0] = *(const uint4*)pb;
            pf.q[1] = *(const uint4*)(pb + 16);
            Frag16 vf[4];
#pragma unroll
            for (int nt = 0; nt < 4; ++nt) {
                const bf16_t* vb = Vts + (nt * 16 + r) * 72 + kk * 32 + hf * 8;
                vf[nt].q[0] = *(const uint4*)vb;
                vf[nt].q[1] = *(const uint4*)(vb + 16);
            }
#pragma unroll
            for (int nt = 0; nt < 4; ++nt)
                O[nt] = wmma_bf16(pf, vf[nt], O[nt]);
        }
        __syncthreads();
    }

    // normalize and write context into Ca[:, h*64 + n] (stride DAUG)
#pragma unroll
    for (int nt = 0; nt < 4; ++nt) {
#pragma unroll
        for (int j = 0; j < 8; ++j) {
            int row = qbase + wv * 16 + hf * 8 + j;
            float o = O[nt][j] / lrun[j];
            Ca[((size_t)(b * SS) + row) * DAUG + h * HDIM + nt * 16 + r] = (bf16_t)o;
        }
    }
}

// ---------------------------------------------------------------------------
// Host launcher
// ---------------------------------------------------------------------------
extern "C" void kernel_launch(void* const* d_in, const int* in_sizes, int n_in,
                              void* d_out, int out_size, void* d_ws, size_t ws_size,
                              hipStream_t stream) {
    (void)in_sizes; (void)n_in; (void)out_size; (void)ws_size;
    const float* hs = (const float*)d_in[0];
    const float* tf = (const float*)d_in[1];
    const float* Wq = (const float*)d_in[2];
    const float* Wk = (const float*)d_in[3];
    const float* Wv = (const float*)d_in[4];
    const float* Wo = (const float*)d_in[5];
    const float* bo = (const float*)d_in[6];
    const float* Wg = (const float*)d_in[7];
    const float* bg = (const float*)d_in[8];
    const float* hp = (const float*)d_in[9];
    const float* tb = (const float*)d_in[10];
    const int*   am = (const int*)d_in[11];
    float* out = (float*)d_out;

    char* ws = (char*)d_ws;
    size_t off = 0;
    auto carve = [&](size_t bytes) -> char* {
        char* p = ws + off;
        off += (bytes + 255) & ~(size_t)255;
        return p;
    };
    bf16_t*   Xa     = (bf16_t*)carve((size_t)MTOT * DAUG * 2);   // [hs | tf] bf16
    bf16_t*   Cau    = (bf16_t*)carve((size_t)MTOT * DAUG * 2);   // [context | tf] bf16
    bf16_t*   WqkvT  = (bf16_t*)carve((size_t)NQKV * DD * 2);
    bf16_t*   WhashT = (bf16_t*)carve((size_t)NHB * DAUG * 2);
    bf16_t*   WgT    = (bf16_t*)carve((size_t)DD * DAUG * 2);
    bf16_t*   WoT    = (bf16_t*)carve((size_t)DD * DD * 2);
    bf16_t*   QKVb   = (bf16_t*)carve((size_t)MTOT * NQKV * 2);
    float*    proj   = (float*)carve((size_t)MTOT * NHB * 4);
    unsigned* codes  = (unsigned*)carve((size_t)MTOT * CWORDS * 4);
    unsigned* keepw  = (unsigned*)carve((size_t)BB * SS * KWORDS * 4);
    bf16_t*   C2     = (bf16_t*)carve((size_t)MTOT * DD * 2);

    const int T = 256;
    auto grid1 = [](long n, int t) { return (unsigned)((n + t - 1) / t); };

    // 1) inputs -> bf16 augmented matrices; weights -> transposed bf16
    prep_x_kernel   <<<grid1((long)MTOT * DAUG, T), T, 0, stream>>>(hs, tf, Xa, Cau);
    pack_wqkv_kernel<<<grid1((long)NQKV * DD, T), T, 0, stream>>>(Wq, Wk, Wv, WqkvT);
    pack_whash_kernel<<<grid1((long)NHB * DAUG, T), T, 0, stream>>>(hp, WhashT);
    pack_wt_kernel  <<<grid1((long)DD * DAUG, T), T, 0, stream>>>(Wg, WgT, DD, DAUG);
    pack_wt_kernel  <<<grid1((long)DD * DD, T), T, 0, stream>>>(Wo, WoT, DD, DD);

    // 2) fused QKV GEMM: [4096x768] x [768x2304]
    gemm_bf16_wmma<<<dim3(MTOT / 128, NQKV / 64), 256, 0, stream>>>(
        Xa, DAUG, WqkvT, DD, DD, 0, nullptr, nullptr, nullptr, QKVb, NQKV, nullptr, 0);

    // 3) LSH projection GEMM: [4096x896] x [896x512] (+ topology bias)
    gemm_bf16_wmma<<<dim3(MTOT / 128, NHB / 64), 256, 0, stream>>>(
        Xa, DAUG, WhashT, DAUG, DAUG, 1, tb, nullptr, nullptr, nullptr, 0, proj, NHB);

    // 4) sign -> packed codes; popcount -> keep bitmask (incl. pad mask)
    pack_codes_kernel<<<grid1((long)MTOT * CWORDS, T), T, 0, stream>>>(proj, codes);
    build_mask_kernel<<<grid1((long)BB * SS * KWORDS, T), T, 0, stream>>>(codes, am, keepw);

    // 5) flash attention with LSH mask; writes context into Cau[:, 0:768]
    attn_kernel<<<dim3(SS / 64, HH, BB), 128, 0, stream>>>(QKVb, keepw, Cau);

    // 6) gate GEMM + sigmoid blend epilogue -> C2
    gemm_bf16_wmma<<<dim3(MTOT / 128, DD / 64), 256, 0, stream>>>(
        Cau, DAUG, WgT, DAUG, DAUG, 2, bg, Cau, hs, C2, DD, nullptr, 0);

    // 7) output GEMM + bias -> f32 d_out
    gemm_bf16_wmma<<<dim3(MTOT / 128, DD / 64), 256, 0, stream>>>(
        C2, DD, WoT, DD, DD, 3, bo, nullptr, nullptr, nullptr, 0, out, DD);
}
